// MHSA_43645457662493
// MI455X (gfx1250) — matrix operand order repaired and checked by reading, not compiled, not run on hardware
//
#include <hip/hip_runtime.h>

typedef __attribute__((ext_vector_type(16))) __bf16 v16bf;
typedef __attribute__((ext_vector_type(8)))  __bf16 v8bf;
typedef __attribute__((ext_vector_type(4)))  __bf16 v4bf;
typedef __attribute__((ext_vector_type(8)))  float  v8f;
typedef __attribute__((ext_vector_type(4)))  int    v4i;

#define B_ 2
#define S_ 2048
#define E_ 1024
#define H_ 16
#define D_ 64

// Async global->LDS staging (CDNA5 GLOBAL_LOAD_ASYNC_TO_LDS_B128, ASYNCcnt),
// guarded so we fall back to the sync copy if this toolchain lacks the builtin.
// Builtin signature (from clang diagnostics): param0 = AS1 int4* (global src),
// param1 = AS3 int4* (LDS dst), then two int immediates (offset, cpol).
#if defined(__gfx1250__) && __has_builtin(__builtin_amdgcn_global_load_async_to_lds_b128) && \
    __has_builtin(__builtin_amdgcn_s_wait_asynccnt)
#define USE_ASYNC_LDS 1
#define AS1P(p) ((__attribute__((address_space(1))) v4i*)(p))
#define AS3P(p) ((__attribute__((address_space(3))) v4i*)(p))
#else
#define USE_ASYNC_LDS 0
#endif

// ---------------------------------------------------------------------------
// WMMA helper: D = A(16x32 bf16) * B(32x16 bf16) + C(16x16 f32)
// ---------------------------------------------------------------------------
static __device__ __forceinline__ v8f wmma_bf16(v16bf a, v16bf b, v8f c) {
  return __builtin_amdgcn_wmma_f32_16x16x32_bf16(false, a, false, b, (short)0, c,
                                                 false, false);
}

// A-matrix 16x32 bf16 fragment: lanes 0-15 hold row M=lane with
// K = {0..15}; lanes 16-31 hold row M=lane-16 with K = {16..31}.
static __device__ __forceinline__ v16bf load_a_frag(const __bf16* base, int stride,
                                                    int lane) {
  const int row = lane & 15;
  const int k0  = (lane >> 4) << 4;  // 0 or 16
  const __bf16* p = base + (size_t)row * stride + k0;
  v8bf lo = *(const v8bf*)(p);
  v8bf hi = *(const v8bf*)(p + 8);
  v16bf r;
#pragma unroll
  for (int i = 0; i < 8; ++i) { r[i] = lo[i]; r[i + 8] = hi[i]; }
  return r;
}

// B-matrix 32x16 bf16 fragment: lane n holds column N=n&15, K-values
// 16*(n>>4) .. +15 packed 2-per-VGPR => one contiguous 32-byte load per lane.
static __device__ __forceinline__ v16bf load_b_frag(const __bf16* base, int stride,
                                                    int lane) {
  const int col = lane & 15;
  const int k0  = (lane >> 4) << 4;  // 0 or 16
  return *(const v16bf*)(base + (size_t)col * stride + k0);
}

// Reductions across the 16-lane halves (wave32): masks 1,2,4,8 never cross
// the half boundary; each half holds a distinct output row.
static __device__ __forceinline__ float hred_sum(float v) {
  v += __shfl_xor(v, 1, 32);
  v += __shfl_xor(v, 2, 32);
  v += __shfl_xor(v, 4, 32);
  v += __shfl_xor(v, 8, 32);
  return v;
}
static __device__ __forceinline__ float hred_max(float v) {
  v = fmaxf(v, __shfl_xor(v, 1, 32));
  v = fmaxf(v, __shfl_xor(v, 2, 32));
  v = fmaxf(v, __shfl_xor(v, 4, 32));
  v = fmaxf(v, __shfl_xor(v, 8, 32));
  return v;
}

// ---------------------------------------------------------------------------
// f32 -> bf16 conversion (x and the three weight matrices), vectorized x4
// ---------------------------------------------------------------------------
__global__ __launch_bounds__(256) void cvt_f32_bf16(const float* __restrict__ in,
                                                    __bf16* __restrict__ out,
                                                    int n4) {
  int i = blockIdx.x * blockDim.x + threadIdx.x;
  if (i < n4) {
    float4 v = ((const float4*)in)[i];
    v4bf o = {(__bf16)v.x, (__bf16)v.y, (__bf16)v.z, (__bf16)v.w};
    ((v4bf*)out)[i] = o;
  }
}

// ---------------------------------------------------------------------------
// Fused QKV projection + per-head L2 norm (Q,K) + V transpose.
// grid = (32, 16, 3): x => 128-row slab, y => head, z => {Q,K,V}.
// Each wave computes one 16x64 tile = one head slice, full K=1024 reduction,
// so the L2 norm over D=64 is done entirely in-register.
// ---------------------------------------------------------------------------
__global__ __launch_bounds__(256) void mhsa_qkv_proj(
    const __bf16* __restrict__ xb, const __bf16* __restrict__ Wqb,
    const __bf16* __restrict__ Wkb, const __bf16* __restrict__ Wvb,
    __bf16* __restrict__ Qg, __bf16* __restrict__ Kg, __bf16* __restrict__ Vtg) {
  const int lane  = threadIdx.x & 31;
  const int wave  = threadIdx.x >> 5;
  const int mbase = blockIdx.x * 128 + wave * 16;  // row in [0, B*S)
  const int h     = blockIdx.y;
  const int z     = blockIdx.z;
  const __bf16* W = (z == 0) ? Wqb : (z == 1) ? Wkb : Wvb;

  const v8f zero8 = {0.f, 0.f, 0.f, 0.f, 0.f, 0.f, 0.f, 0.f};
  v8f acc[4];
#pragma unroll
  for (int c = 0; c < 4; ++c) acc[c] = zero8;

  const __bf16* arow = xb + (size_t)mbase * E_;
  const __bf16* wrow = W + (size_t)h * D_ * E_;

  for (int k = 0; k < E_; k += 32) {
    __builtin_prefetch(arow + (size_t)(lane & 15) * E_ + k + 256, 0, 1);
    v16bf a = load_a_frag(arow + k, E_, lane);
#pragma unroll
    for (int c = 0; c < 4; ++c) {
      v16bf b = load_b_frag(wrow + (size_t)(c * 16) * E_ + k, E_, lane);
      acc[c] = wmma_bf16(a, b, acc[c]);
    }
  }

  const int hl = lane >> 4, ln = lane & 15;
  float inv[8];
  if (z < 2) {  // L2 normalize q,k rows (torch F.normalize, eps=1e-12)
#pragma unroll
    for (int r = 0; r < 8; ++r) {
      float s = 0.f;
#pragma unroll
      for (int c = 0; c < 4; ++c) s += acc[c][r] * acc[c][r];
      s = hred_sum(s);
      inv[r] = 1.0f / fmaxf(sqrtf(s), 1e-12f);
    }
  } else {
#pragma unroll
    for (int r = 0; r < 8; ++r) inv[r] = 1.0f;
  }

  const int b0   = mbase >> 11;        // / S_
  const int srow = mbase & (S_ - 1);   // % S_
  const size_t bh = (size_t)(b0 * H_ + h);

  if (z == 2) {  // V stored transposed: [B,H,D,S] so P*V B-frags are contiguous
    __bf16* dst = Vtg + bh * (size_t)(D_ * S_);
#pragma unroll
    for (int r = 0; r < 8; ++r) {
      const int m = r + 8 * hl;  // C layout: VGPR r holds rows r / r+8
#pragma unroll
      for (int c = 0; c < 4; ++c)
        dst[(size_t)(c * 16 + ln) * S_ + (srow + m)] = (__bf16)acc[c][r];
    }
  } else {       // Q/K stored [B,H,S,D]
    __bf16* dst = ((z == 0) ? Qg : Kg) + bh * (size_t)(S_ * D_);
#pragma unroll
    for (int r = 0; r < 8; ++r) {
      const int m = r + 8 * hl;
#pragma unroll
      for (int c = 0; c < 4; ++c)
        dst[(size_t)(srow + m) * D_ + c * 16 + ln] = (__bf16)(acc[c][r] * inv[r]);
    }
  }
}

// ---------------------------------------------------------------------------
// Flash attention: grid = (S/128, H, B); 8 waves, each owns 16 query rows.
// K/Vt chunks (64 keys) staged in LDS (async DMA when available), shared by
// all 8 waves.
// ---------------------------------------------------------------------------
__global__ __launch_bounds__(256) void mhsa_flash_attn(
    const __bf16* __restrict__ Qg, const __bf16* __restrict__ Kg,
    const __bf16* __restrict__ Vtg, const float* __restrict__ g_scale,
    float* __restrict__ Og) {
  __shared__ __align__(32) __bf16 Ksh[64 * 64];       // [key][d]
  __shared__ __align__(32) __bf16 Vsh[64 * 64];       // [d][key]
  __shared__ __align__(32) __bf16 Psh[8 * 16 * 64];   // per-wave P staging

  const int lane = threadIdx.x & 31;
  const int wave = threadIdx.x >> 5;
  const int hl = lane >> 4, ln = lane & 15;
  const int qbase = blockIdx.x * 128 + wave * 16;
  const int h  = blockIdx.y;
  const int bz = blockIdx.z;
  const float gs = g_scale[0];

  const size_t bh = (size_t)(bz * H_ + h);
  const __bf16* Qp  = Qg  + bh * (size_t)(S_ * D_);
  const __bf16* Kp  = Kg  + bh * (size_t)(S_ * D_);
  const __bf16* Vtp = Vtg + bh * (size_t)(D_ * S_);

  // Q fragments for this wave's 16 rows, resident for the whole kernel.
  v16bf qf[2];
#pragma unroll
  for (int f = 0; f < 2; ++f)
    qf[f] = load_a_frag(Qp + (size_t)qbase * D_ + f * 32, D_, lane);

  const v8f zero8 = {0.f, 0.f, 0.f, 0.f, 0.f, 0.f, 0.f, 0.f};
  v8f o[4];
#pragma unroll
  for (int c = 0; c < 4; ++c) o[c] = zero8;
  float rowmax[8], rowsum[8];
#pragma unroll
  for (int r = 0; r < 8; ++r) { rowmax[r] = -1e30f; rowsum[r] = 0.f; }

  __bf16* Pw = Psh + wave * (16 * 64);

  for (int kb = 0; kb < S_; kb += 64) {
    __syncthreads();  // previous iteration's LDS reads complete
    {                 // cooperative stage: 64x64 bf16 K tile + Vt tile
      const int t   = threadIdx.x;
      const int row = t >> 2;
      const int ch  = (t & 3) * 16;
#if USE_ASYNC_LDS
      // Direct global->LDS DMA, 16B per lane, tracked by ASYNCcnt.
      __builtin_amdgcn_global_load_async_to_lds_b128(
          AS1P(Kp + (size_t)(kb + row) * D_ + ch), AS3P(&Ksh[row * 64 + ch]),
          0, 0);
      __builtin_amdgcn_global_load_async_to_lds_b128(
          AS1P(Vtp + (size_t)row * S_ + kb + ch), AS3P(&Vsh[row * 64 + ch]),
          0, 0);
      __builtin_amdgcn_s_wait_asynccnt(0);
#else
      *(v16bf*)&Ksh[row * 64 + ch] =
          *(const v16bf*)(Kp + (size_t)(kb + row) * D_ + ch);
      *(v16bf*)&Vsh[row * 64 + ch] =
          *(const v16bf*)(Vtp + (size_t)row * S_ + kb + ch);
#endif
    }
    __syncthreads();

    // scores S = Q * K^T for 16 q-rows x 64 keys
    v8f sacc[4];
#pragma unroll
    for (int c = 0; c < 4; ++c) sacc[c] = zero8;
#pragma unroll
    for (int c = 0; c < 4; ++c)
#pragma unroll
      for (int f = 0; f < 2; ++f) {
        v16bf b = load_b_frag(&Ksh[(c * 16) * 64 + f * 32], 64, lane);
        sacc[c] = wmma_bf16(qf[f], b, sacc[c]);
      }

    // online softmax (gs > 0 so max commutes with the scale)
    float pv[4][8];
#pragma unroll
    for (int r = 0; r < 8; ++r) {
      float cm = sacc[0][r];
      cm = fmaxf(cm, sacc[1][r]);
      cm = fmaxf(cm, sacc[2][r]);
      cm = fmaxf(cm, sacc[3][r]);
      cm = hred_max(cm) * gs;
      const float nm   = fmaxf(rowmax[r], cm);
      const float corr = __expf(rowmax[r] - nm);
      rowmax[r] = nm;
      float rs = 0.f;
#pragma unroll
      for (int c = 0; c < 4; ++c) {
        const float p = __expf(sacc[c][r] * gs - nm);
        pv[c][r] = p;
        rs += p;
      }
      rowsum[r] = rowsum[r] * corr + hred_sum(rs);
#pragma unroll
      for (int c = 0; c < 4; ++c) o[c][r] *= corr;
    }

    // C-layout -> A-layout via per-wave LDS bounce (wave-local, DS in-order)
#pragma unroll
    for (int r = 0; r < 8; ++r)
#pragma unroll
      for (int c = 0; c < 4; ++c)
        Pw[(r + 8 * hl) * 64 + c * 16 + ln] = (__bf16)pv[c][r];

    // O += P * V
#pragma unroll
    for (int f = 0; f < 2; ++f) {
      v16bf a = load_a_frag(Pw + f * 32, 64, lane);
#pragma unroll
      for (int c = 0; c < 4; ++c) {
        v16bf b = load_b_frag(&Vsh[(c * 16) * 64 + f * 32], 64, lane);
        o[c] = wmma_bf16(a, b, o[c]);
      }
    }
  }

  // epilogue: divide by softmax denominator, write f32 [B,S,E]
#pragma unroll
  for (int r = 0; r < 8; ++r) {
    const float inv = 1.0f / rowsum[r];
    const int m = r + 8 * hl;
    const size_t rowoff = ((size_t)(bz * S_ + qbase + m)) * E_ + (size_t)h * D_;
#pragma unroll
    for (int c = 0; c < 4; ++c) Og[rowoff + c * 16 + ln] = o[c][r] * inv;
  }
}

// ---------------------------------------------------------------------------
extern "C" void kernel_launch(void* const* d_in, const int* in_sizes, int n_in,
                              void* d_out, int out_size, void* d_ws, size_t ws_size,
                              hipStream_t stream) {
  (void)in_sizes; (void)n_in; (void)out_size; (void)ws_size;
  const float* x  = (const float*)d_in[0];
  const float* Wq = (const float*)d_in[1];
  const float* Wk = (const float*)d_in[2];
  const float* Wv = (const float*)d_in[3];
  const float* gs = (const float*)d_in[4];
  float* out = (float*)d_out;

  const size_t nX = (size_t)B_ * S_ * E_;       // 4,194,304
  const size_t nW = (size_t)E_ * E_;            // 1,048,576
  const size_t nH = (size_t)B_ * H_ * S_ * D_;  // 4,194,304

  char* ws = (char*)d_ws;
  __bf16* xb  = (__bf16*)ws;              ws += nX * 2;
  __bf16* Wqb = (__bf16*)ws;              ws += nW * 2;
  __bf16* Wkb = (__bf16*)ws;              ws += nW * 2;
  __bf16* Wvb = (__bf16*)ws;              ws += nW * 2;
  __bf16* Qb  = (__bf16*)ws;              ws += nH * 2;
  __bf16* Kb  = (__bf16*)ws;              ws += nH * 2;
  __bf16* Vtb = (__bf16*)ws;              // transposed V: [B,H,D,S]

  cvt_f32_bf16<<<(int)(nX / 4 / 256), 256, 0, stream>>>(x, xb, (int)(nX / 4));
  cvt_f32_bf16<<<(int)(nW / 4 / 256), 256, 0, stream>>>(Wq, Wqb, (int)(nW / 4));
  cvt_f32_bf16<<<(int)(nW / 4 / 256), 256, 0, stream>>>(Wk, Wkb, (int)(nW / 4));
  cvt_f32_bf16<<<(int)(nW / 4 / 256), 256, 0, stream>>>(Wv, Wvb, (int)(nW / 4));

  mhsa_qkv_proj<<<dim3((B_ * S_) / 128, H_, 3), 256, 0, stream>>>(
      xb, Wqb, Wkb, Wvb, Qb, Kb, Vtb);

  mhsa_flash_attn<<<dim3(S_ / 128, H_, B_), 256, 0, stream>>>(
      Qb, Kb, Vtb, gs, out);
}
